// MolGraphEncoder_80719615361109
// MI455X (gfx1250) — matrix operand order repaired and checked by reading, not compiled, not run on hardware
//
#include <hip/hip_runtime.h>

#define NA    65536
#define NB    131072
#define H     256
#define NMOL  4096
#define ADIM  39
#define BDIM  11
#define KNBR  6
#define TILE_M 32

typedef __bf16 bf16_t;
typedef __attribute__((ext_vector_type(16))) __bf16 v16bf;
typedef __attribute__((ext_vector_type(8)))  __bf16 v8bf;
typedef __attribute__((ext_vector_type(8)))  float  v8f;
typedef int v4i_vs __attribute__((vector_size(16)));   // matches async builtin proto

#ifndef __has_builtin
#define __has_builtin(x) 0
#endif
#if __has_builtin(__builtin_amdgcn_global_load_async_to_lds_b128) && \
    __has_builtin(__builtin_amdgcn_s_wait_asynccnt)
#define ASYNC_OK 1
#else
#define ASYNC_OK 0
#endif

#if ASYNC_OK
#define WAIT_ASYNC(n) __builtin_amdgcn_s_wait_asynccnt(n)
#else
#define WAIT_ASYNC(n) ((void)0)
#endif

// ---------------------------------------------------------------------------
// WMMA fragment helpers (wave32, v_wmma_f32_16x16x32_bf16)
// A (16x32): lanes 0-15: M=lane, elems-> K{k0..k0+7, k0+16..k0+23};
//            lanes 16-31: M=lane-16, K{k0+8..15, k0+24..31}
// B (32x16) from N-major weight: lanes 0-15: N=n0+lane, K{k0..k0+7, k0+16..k0+23};
//            lanes 16-31: K{k0+8..15, k0+24..31}
// C/D: VGPR r: lanes 0-15 -> M=r, N=lane; lanes 16-31 -> M=8+r, N=lane-16
// ---------------------------------------------------------------------------

static __device__ __forceinline__ v16bf load_frag_a_lds(const bf16_t* __restrict__ tile,
                                                        int rlocal, int k0, int lane) {
  const int half = lane >> 4;
  const bf16_t* p = tile + rlocal * H + k0 + half * 8;
  v8bf lo = *(const v8bf*)(p);
  v8bf hi = *(const v8bf*)(p + 16);
  v16bf a;
#pragma unroll
  for (int i = 0; i < 8; ++i) { a[i] = lo[i]; a[i + 8] = hi[i]; }
  return a;
}

static __device__ __forceinline__ v16bf load_frag_b(const bf16_t* __restrict__ wt,
                                                    int n0, int k0, int lane) {
  const int half = lane >> 4;
  const int n = n0 + (lane & 15);
  const bf16_t* p = wt + (size_t)n * H + k0 + half * 8;
  v8bf lo = *(const v8bf*)(p);
  v8bf hi = *(const v8bf*)(p + 16);
  v16bf b;
#pragma unroll
  for (int i = 0; i < 8; ++i) { b[i] = lo[i]; b[i + 8] = hi[i]; }
  return b;
}

static __device__ __forceinline__ v8f wmma_bf16(v16bf a, v16bf b, v8f c) {
  return __builtin_amdgcn_wmma_f32_16x16x32_bf16(false, a, false, b, (short)0, c,
                                                 false, false);
}

// ---------------------------------------------------------------------------
// Stage one 32-row x 256-col bf16 activation tile into LDS.
// Thread t stages 64B of tile row (t>>3), cols [(t&7)*32, +32).
// Async path: 4 GLOBAL_LOAD_ASYNC_TO_LDS_B128 per wave (ASYNCcnt += 4).
// ---------------------------------------------------------------------------
static __device__ __forceinline__ void stage_tile(bf16_t* lds_tile,
                                                  const bf16_t* __restrict__ src,
                                                  int row_global, int tid) {
  const int r = tid >> 3;
  const int c = (tid & 7) * 32;
  const bf16_t* g = src + (size_t)row_global * H + c;
  bf16_t* l = lds_tile + r * H + c;
#if ASYNC_OK
#pragma unroll
  for (int i = 0; i < 4; ++i)
    __builtin_amdgcn_global_load_async_to_lds_b128(
        (v4i_vs*)(g + i * 8), (v4i_vs*)(l + i * 8), 0, 0);
#else
#pragma unroll
  for (int i = 0; i < 4; ++i)
    *(v8bf*)(l + i * 8) = *(const v8bf*)(g + i * 8);
#endif
}

// ---------------------------------------------------------------------------
// c[0..3] += tile_row @ wt for one K=256 stream; A (LDS) and B (global, L2-hot)
// fragments are double-buffered so loads for k+32 are in flight across the
// WMMAs of k.
// ---------------------------------------------------------------------------
static __device__ __forceinline__ void compute_stream(v8f c[4],
                                                      const bf16_t* __restrict__ tile,
                                                      int rlocal,
                                                      const bf16_t* __restrict__ wt,
                                                      int n0, int lane) {
  v16bf acur = load_frag_a_lds(tile, rlocal, 0, lane);
  v16bf bcur[4];
#pragma unroll
  for (int nt = 0; nt < 4; ++nt) bcur[nt] = load_frag_b(wt, n0 + nt * 16, 0, lane);
#pragma unroll
  for (int k0 = 0; k0 < H; k0 += 32) {
    v16bf anxt = acur;
    v16bf bnxt[4] = {bcur[0], bcur[1], bcur[2], bcur[3]};
    if (k0 + 32 < H) {
      anxt = load_frag_a_lds(tile, rlocal, k0 + 32, lane);
#pragma unroll
      for (int nt = 0; nt < 4; ++nt) bnxt[nt] = load_frag_b(wt, n0 + nt * 16, k0 + 32, lane);
    }
#pragma unroll
    for (int nt = 0; nt < 4; ++nt) c[nt] = wmma_bf16(acur, bcur[nt], c[nt]);
    acur = anxt;
#pragma unroll
    for (int nt = 0; nt < 4; ++nt) bcur[nt] = bnxt[nt];
  }
}

// Same A-stream feeding two weight matrices / accumulator sets (readout).
static __device__ __forceinline__ void compute_dual(v8f c1[4], v8f c2[4],
                                                    const bf16_t* __restrict__ tile,
                                                    int rlocal,
                                                    const bf16_t* __restrict__ wt1,
                                                    const bf16_t* __restrict__ wt2,
                                                    int n0, int lane) {
  v16bf acur = load_frag_a_lds(tile, rlocal, 0, lane);
#pragma unroll
  for (int k0 = 0; k0 < H; k0 += 32) {
    v16bf anxt = acur;
    if (k0 + 32 < H) anxt = load_frag_a_lds(tile, rlocal, k0 + 32, lane);
#pragma unroll
    for (int nt = 0; nt < 4; ++nt) {
      c1[nt] = wmma_bf16(acur, load_frag_b(wt1, n0 + nt * 16, k0, lane), c1[nt]);
      c2[nt] = wmma_bf16(acur, load_frag_b(wt2, n0 + nt * 16, k0, lane), c2[nt]);
    }
    acur = anxt;
  }
}

static __device__ __forceinline__ float sigmoidf(float x) {
  return 1.0f / (1.0f + __expf(-x));
}

// ---------------------------------------------------------------------------
// Weight convert+transpose: w[k][n] f32 (256x256) -> wt[n][k] bf16
// ---------------------------------------------------------------------------
__global__ void __launch_bounds__(256)
convert_wt_kernel(const float* __restrict__ w, bf16_t* __restrict__ wt) {
  __shared__ bf16_t tile[32][33];
  const int tx = threadIdx.x & 31;
  const int ty = threadIdx.x >> 5;
  const int nb = blockIdx.x * 32;
  const int kb = blockIdx.y * 32;
#pragma unroll
  for (int i = 0; i < 32; i += 8)
    tile[ty + i][tx] = (bf16_t)w[(size_t)(kb + ty + i) * H + nb + tx];
  __syncthreads();
#pragma unroll
  for (int i = 0; i < 32; i += 8)
    wt[(size_t)(nb + ty + i) * H + kb + tx] = tile[tx][ty + i];
}

// ---------------------------------------------------------------------------
// Embedding: out[row] = relu(feat[row] @ E + Eb), bf16 out. One block per row.
// ---------------------------------------------------------------------------
__global__ void __launch_bounds__(256)
embed_kernel(const float* __restrict__ feat, int fd,
             const float* __restrict__ E, const float* __restrict__ Eb,
             bf16_t* __restrict__ out) {
  __shared__ float sf[64];
  const int row = blockIdx.x;
  const int t = threadIdx.x;
  if (t < fd) sf[t] = feat[(size_t)row * fd + t];
  __syncthreads();
  float acc = Eb[t];
  for (int d = 0; d < fd; ++d) acc = fmaf(sf[d], E[(size_t)d * H + t], acc);
  out[(size_t)row * H + t] = (bf16_t)fmaxf(acc, 0.0f);
}

// ---------------------------------------------------------------------------
// Message: msg[i] = sum_k sigmoid(bh[abg[i,k]]) * ah[aag[i,k]]. Block per atom.
// ---------------------------------------------------------------------------
__global__ void __launch_bounds__(256)
message_kernel(const bf16_t* __restrict__ ah, const bf16_t* __restrict__ bh,
               const int* __restrict__ aag, const int* __restrict__ abg,
               bf16_t* __restrict__ msg) {
  const int atom = blockIdx.x;
  const int t = threadIdx.x;
  float acc = 0.0f;
#pragma unroll
  for (int k = 0; k < KNBR; ++k) {
    const int ai = aag[atom * KNBR + k];
    const int bi = abg[atom * KNBR + k];
    const float a = (float)ah[(size_t)ai * H + t];
    const float g = sigmoidf((float)bh[(size_t)bi * H + t]);
    acc = fmaf(g, a, acc);
  }
  msg[(size_t)atom * H + t] = (bf16_t)acc;
}

// ---------------------------------------------------------------------------
// Atom update: out = relu(ah@Ws + msg@Wn + bs + bn)
// Block = 32 M rows x full N; 8 waves as 2(M) x 4(N); 2 async-staged streams.
// ---------------------------------------------------------------------------
__global__ void __launch_bounds__(256)
atom_update_kernel(const bf16_t* __restrict__ ah, const bf16_t* __restrict__ msg,
                   const bf16_t* __restrict__ wst, const bf16_t* __restrict__ wnt,
                   const float* __restrict__ bs, const float* __restrict__ bn,
                   bf16_t* __restrict__ out) {
  __shared__ __align__(16) bf16_t tile[2][TILE_M * H];
  const int tid = threadIdx.x;
  const int lane = tid & 31;
  const int wave = tid >> 5;
  const int mblock = blockIdx.x * TILE_M;
  const int m0 = mblock + (wave >> 2) * 16;
  const int n0 = (wave & 3) * 64;
  const int rstage = tid >> 3;

  stage_tile(tile[0], ah,  mblock + rstage, tid);
  stage_tile(tile[1], msg, mblock + rstage, tid);

  v8f c[4] = {v8f{}, v8f{}, v8f{}, v8f{}};
  const int rlocal = (wave >> 2) * 16 + (lane & 15);

  WAIT_ASYNC(4);           // stream-0 tile resident (stream-1 still in flight)
  __syncthreads();
  compute_stream(c, tile[0], rlocal, wst, n0, lane);
  WAIT_ASYNC(0);
  __syncthreads();
  compute_stream(c, tile[1], rlocal, wnt, n0, lane);

  const int half = lane >> 4;
#pragma unroll
  for (int nt = 0; nt < 4; ++nt) {
    const int n = n0 + nt * 16 + (lane & 15);
    const float bias = bs[n] + bn[n];
#pragma unroll
    for (int r = 0; r < 8; ++r) {
      const int m = m0 + half * 8 + r;
      out[(size_t)m * H + n] = (bf16_t)fmaxf(c[nt][r] + bias, 0.0f);
    }
  }
}

// ---------------------------------------------------------------------------
// Bond update: out = relu(bh@Wb + bb + ah[x]@Wx + ah[y]@Wy)
// 3 streams; gathered rows resolved at staging time; double-buffered LDS.
// ---------------------------------------------------------------------------
__global__ void __launch_bounds__(256)
bond_update_kernel(const bf16_t* __restrict__ bh, const bf16_t* __restrict__ ah,
                   const int* __restrict__ bag,
                   const bf16_t* __restrict__ wbt, const bf16_t* __restrict__ wxt,
                   const bf16_t* __restrict__ wyt, const float* __restrict__ bb,
                   bf16_t* __restrict__ out) {
  __shared__ __align__(16) bf16_t tile[2][TILE_M * H];
  const int tid = threadIdx.x;
  const int lane = tid & 31;
  const int wave = tid >> 5;
  const int mblock = blockIdx.x * TILE_M;
  const int m0 = mblock + (wave >> 2) * 16;
  const int n0 = (wave & 3) * 64;
  const int rstage = tid >> 3;
  const int xs = bag[(mblock + rstage) * 2 + 0];
  const int ys = bag[(mblock + rstage) * 2 + 1];

  stage_tile(tile[0], bh, mblock + rstage, tid);
  stage_tile(tile[1], ah, xs, tid);

  v8f c[4] = {v8f{}, v8f{}, v8f{}, v8f{}};
  const int rlocal = (wave >> 2) * 16 + (lane & 15);

  WAIT_ASYNC(4);
  __syncthreads();
  compute_stream(c, tile[0], rlocal, wbt, n0, lane);
  WAIT_ASYNC(0);
  __syncthreads();                    // x-tile ready; everyone done with tile[0]
  stage_tile(tile[0], ah, ys, tid);   // overlap y-gather with x compute
  compute_stream(c, tile[1], rlocal, wxt, n0, lane);
  WAIT_ASYNC(0);
  __syncthreads();
  compute_stream(c, tile[0], rlocal, wyt, n0, lane);

  const int half = lane >> 4;
#pragma unroll
  for (int nt = 0; nt < 4; ++nt) {
    const int n = n0 + nt * 16 + (lane & 15);
    const float bias = bb[n];
#pragma unroll
    for (int r = 0; r < 8; ++r) {
      const int mm = m0 + half * 8 + r;
      out[(size_t)mm * H + n] = (bf16_t)fmaxf(c[nt][r] + bias, 0.0f);
    }
  }
}

// ---------------------------------------------------------------------------
// Readout:
//   gate  = sigmoid(bh@U + Ub + ah[x]@V + Vb + ah[y]@Wg + Wgb)
//   gated = gate * (bh@A + Ab)     (bf16 out)
// bh tile feeds both U and A accumulators (A-fragment reuse).
// ---------------------------------------------------------------------------
__global__ void __launch_bounds__(256)
final_gemm_kernel(const bf16_t* __restrict__ bh, const bf16_t* __restrict__ ah,
                  const int* __restrict__ bag,
                  const bf16_t* __restrict__ ut, const bf16_t* __restrict__ vt,
                  const bf16_t* __restrict__ wgt, const bf16_t* __restrict__ at,
                  const float* __restrict__ ub, const float* __restrict__ vb,
                  const float* __restrict__ wgb, const float* __restrict__ ab,
                  bf16_t* __restrict__ gated) {
  __shared__ __align__(16) bf16_t tile[2][TILE_M * H];
  const int tid = threadIdx.x;
  const int lane = tid & 31;
  const int wave = tid >> 5;
  const int mblock = blockIdx.x * TILE_M;
  const int m0 = mblock + (wave >> 2) * 16;
  const int n0 = (wave & 3) * 64;
  const int rstage = tid >> 3;
  const int xs = bag[(mblock + rstage) * 2 + 0];
  const int ys = bag[(mblock + rstage) * 2 + 1];

  stage_tile(tile[0], bh, mblock + rstage, tid);
  stage_tile(tile[1], ah, xs, tid);

  v8f c1[4] = {v8f{}, v8f{}, v8f{}, v8f{}};
  v8f c2[4] = {v8f{}, v8f{}, v8f{}, v8f{}};
  const int rlocal = (wave >> 2) * 16 + (lane & 15);

  WAIT_ASYNC(4);
  __syncthreads();
  compute_dual(c1, c2, tile[0], rlocal, ut, at, n0, lane);
  WAIT_ASYNC(0);
  __syncthreads();
  stage_tile(tile[0], ah, ys, tid);
  compute_stream(c1, tile[1], rlocal, vt, n0, lane);
  WAIT_ASYNC(0);
  __syncthreads();
  compute_stream(c1, tile[0], rlocal, wgt, n0, lane);

  const int half = lane >> 4;
#pragma unroll
  for (int nt = 0; nt < 4; ++nt) {
    const int n = n0 + nt * 16 + (lane & 15);
    const float b1 = ub[n] + vb[n] + wgb[n];
    const float b2 = ab[n];
#pragma unroll
    for (int r = 0; r < 8; ++r) {
      const int mm = m0 + half * 8 + r;
      const float eg = sigmoidf(c1[nt][r] + b1);
      gated[(size_t)mm * H + n] = (bf16_t)(eg * (c2[nt][r] + b2));
    }
  }
}

// ---------------------------------------------------------------------------
// Deterministic segment sum over sorted bond_scope_ids. Block per molecule.
// Writes every output element (zero-fills empty molecules: no memset needed).
// ---------------------------------------------------------------------------
__global__ void __launch_bounds__(256)
segsum_kernel(const bf16_t* __restrict__ gated, const int* __restrict__ seg,
              float* __restrict__ out) {
  const int mol = blockIdx.x;
  const int t = threadIdx.x;
  int lo = 0, hi = NB;
  while (lo < hi) { int mid = (lo + hi) >> 1; if (seg[mid] < mol) lo = mid + 1; else hi = mid; }
  const int start = lo;
  hi = NB;
  while (lo < hi) { int mid = (lo + hi) >> 1; if (seg[mid] < mol + 1) lo = mid + 1; else hi = mid; }
  const int end = lo;
  float acc = 0.0f;
  for (int b = start; b < end; ++b) acc += (float)gated[(size_t)b * H + t];
  out[(size_t)mol * H + t] = acc;
}

// ---------------------------------------------------------------------------
extern "C" void kernel_launch(void* const* d_in, const int* in_sizes, int n_in,
                              void* d_out, int out_size, void* d_ws, size_t ws_size,
                              hipStream_t stream) {
  (void)in_sizes; (void)n_in; (void)out_size; (void)ws_size;
  const float* af   = (const float*)d_in[0];
  const float* bff  = (const float*)d_in[1];
  const int*   aag  = (const int*)d_in[2];
  const int*   abg  = (const int*)d_in[3];
  const int*   bag  = (const int*)d_in[4];
  const int*   seg  = (const int*)d_in[5];
  const float* Ea   = (const float*)d_in[6];
  const float* Ea_b = (const float*)d_in[7];
  const float* Eb   = (const float*)d_in[8];
  const float* Eb_b = (const float*)d_in[9];
  const float* Ws   = (const float*)d_in[10];
  const float* Ws_b = (const float*)d_in[11];
  const float* Wn   = (const float*)d_in[12];
  const float* Wn_b = (const float*)d_in[13];
  const float* Wb   = (const float*)d_in[14];
  const float* Wb_b = (const float*)d_in[15];
  const float* Wx   = (const float*)d_in[16];
  const float* Wy   = (const float*)d_in[17];
  const float* A    = (const float*)d_in[18];
  const float* A_b  = (const float*)d_in[19];
  const float* U    = (const float*)d_in[20];
  const float* U_b  = (const float*)d_in[21];
  const float* V    = (const float*)d_in[22];
  const float* V_b  = (const float*)d_in[23];
  const float* Wg   = (const float*)d_in[24];
  const float* Wg_b = (const float*)d_in[25];
  float* out = (float*)d_out;

  char* ws = (char*)d_ws;
  size_t off = 0;
  auto alloc = [&](size_t bytes) -> void* {
    void* p = ws + off;
    off += (bytes + 255) & ~(size_t)255;
    return p;
  };
  bf16_t* ah0  = (bf16_t*)alloc((size_t)NA * H * sizeof(bf16_t));
  bf16_t* ah1  = (bf16_t*)alloc((size_t)NA * H * sizeof(bf16_t));
  bf16_t* bh0  = (bf16_t*)alloc((size_t)NB * H * sizeof(bf16_t));
  bf16_t* bh1  = (bf16_t*)alloc((size_t)NB * H * sizeof(bf16_t));
  bf16_t* msgb = (bf16_t*)alloc((size_t)NA * H * sizeof(bf16_t));
  bf16_t* wt   = (bf16_t*)alloc((size_t)19 * H * H * sizeof(bf16_t));
  auto WT = [&](int i) { return wt + (size_t)i * H * H; };
  // slots: 0-2 Ws[l], 3-5 Wn[l], 6-8 Wb[l], 9-11 Wx[l], 12-14 Wy[l],
  //        15 A, 16 U, 17 V, 18 Wg
  const dim3 tgrid(8, 8);
  for (int l = 0; l < 3; ++l) {
    convert_wt_kernel<<<tgrid, 256, 0, stream>>>(Ws + (size_t)l * H * H, WT(0 + l));
    convert_wt_kernel<<<tgrid, 256, 0, stream>>>(Wn + (size_t)l * H * H, WT(3 + l));
    convert_wt_kernel<<<tgrid, 256, 0, stream>>>(Wb + (size_t)l * H * H, WT(6 + l));
    convert_wt_kernel<<<tgrid, 256, 0, stream>>>(Wx + (size_t)l * H * H, WT(9 + l));
    convert_wt_kernel<<<tgrid, 256, 0, stream>>>(Wy + (size_t)l * H * H, WT(12 + l));
  }
  convert_wt_kernel<<<tgrid, 256, 0, stream>>>(A,  WT(15));
  convert_wt_kernel<<<tgrid, 256, 0, stream>>>(U,  WT(16));
  convert_wt_kernel<<<tgrid, 256, 0, stream>>>(V,  WT(17));
  convert_wt_kernel<<<tgrid, 256, 0, stream>>>(Wg, WT(18));

  embed_kernel<<<NA, 256, 0, stream>>>(af,  ADIM, Ea, Ea_b, ah0);
  embed_kernel<<<NB, 256, 0, stream>>>(bff, BDIM, Eb, Eb_b, bh0);

  bf16_t* ahc = ah0; bf16_t* ahn = ah1;
  bf16_t* bhc = bh0; bf16_t* bhn = bh1;
  for (int l = 0; l < 3; ++l) {
    message_kernel<<<NA, 256, 0, stream>>>(ahc, bhc, aag, abg, msgb);
    atom_update_kernel<<<NA / TILE_M, 256, 0, stream>>>(
        ahc, msgb, WT(0 + l), WT(3 + l), Ws_b + l * H, Wn_b + l * H, ahn);
    bond_update_kernel<<<NB / TILE_M, 256, 0, stream>>>(
        bhc, ahn, bag, WT(6 + l), WT(9 + l), WT(12 + l), Wb_b + l * H, bhn);
    bf16_t* t;
    t = ahc; ahc = ahn; ahn = t;
    t = bhc; bhc = bhn; bhn = t;
  }
  // After 3 layers: state in ahc (=ah1), bhc (=bh1); bhn (=bh0) is free scratch
  bf16_t* gated = bhn;
  final_gemm_kernel<<<NB / TILE_M, 256, 0, stream>>>(
      bhc, ahc, bag, WT(16), WT(17), WT(18), WT(15), U_b, V_b, Wg_b, A_b, gated);
  segsum_kernel<<<NMOL, 256, 0, stream>>>(gated, seg, out);
}
